// RGB_HVI_65146063946070
// MI455X (gfx1250) — matrix operand order repaired and checked by reading, not compiled, not run on hardware
//
#include <hip/hip_runtime.h>
#include <hip/hip_bf16.h>
#include <math.h>

typedef __attribute__((ext_vector_type(16))) _Float16 v16h;
typedef __attribute__((ext_vector_type(8)))  _Float16 v8h;
typedef __attribute__((ext_vector_type(8)))  float    v8f;

#define HDIM 512
#define WDIM 512
#define BATCH 16
#define PI_F 3.14159265358979323846f
#define EPS_F 1e-8f

// ---------------------------------------------------------------------------
// Kernel 0: pack conv2 weights into per-lane WMMA B-matrix layout, f16.
// wpack[g][lane][i] : b[i] = w2[n][8*hi + (i&7)][ky][kx] of tap (2g + (i>>3));
//                     (n = lane&15, hi = lane>>4).
// One wave does everything (5 KB output).
// ---------------------------------------------------------------------------
__global__ void pack_w2_kernel(const float* __restrict__ w2, _Float16* __restrict__ wpack)
{
    const int lane = threadIdx.x;     // 32 threads
    const int n = lane & 15;
    const int hi = lane >> 4;
    #pragma unroll
    for (int g = 0; g < 5; ++g) {
        v16h bv;
        #pragma unroll
        for (int i = 0; i < 16; ++i) {
            const int T = 2 * g + (i >> 3);
            const int c = 8 * hi + (i & 7);
            if (T < 9) {
                const int ky = T / 3, kx = T % 3;
                bv[i] = (_Float16)w2[((n * 16 + c) * 3 + ky) * 3 + kx];
            } else {
                bv[i] = (_Float16)0.f;
            }
        }
        *(v16h*)(wpack + ((size_t)g * 32 + lane) * 16) = bv;
    }
}

// ---------------------------------------------------------------------------
// Kernel 1: conv1 (3 -> 16 ch, 3x3 SAME) + ReLU, fp32 VALU, h1 out as f16 NHWC
// ---------------------------------------------------------------------------
__global__ void __launch_bounds__(256)
conv1_kernel(const float* __restrict__ img, const float* __restrict__ w1,
             const float* __restrict__ b1, _Float16* __restrict__ h1)
{
    const int HW = HDIM * WDIM;
    int idx = blockIdx.x * blockDim.x + threadIdx.x;
    if (idx >= BATCH * HW) return;
    int b = idx / HW;
    int p = idx - b * HW;
    int y = p >> 9;
    int x = p & (WDIM - 1);

    float in[3][3][3];
    #pragma unroll
    for (int c = 0; c < 3; ++c)
        #pragma unroll
        for (int dy = 0; dy < 3; ++dy)
            #pragma unroll
            for (int dx = 0; dx < 3; ++dx) {
                int yy = y + dy - 1, xx = x + dx - 1;
                float v = 0.f;
                if (yy >= 0 && yy < HDIM && xx >= 0 && xx < WDIM)
                    v = img[(((size_t)b * 3 + c) * HDIM + yy) * WDIM + xx];
                in[c][dy][dx] = v;
            }

    v16h ov;
    #pragma unroll
    for (int o = 0; o < 16; ++o) {
        float acc = b1[o];
        #pragma unroll
        for (int c = 0; c < 3; ++c)
            #pragma unroll
            for (int ky = 0; ky < 3; ++ky)
                #pragma unroll
                for (int kx = 0; kx < 3; ++kx)
                    acc = fmaf(w1[((o * 3 + c) * 3 + ky) * 3 + kx], in[c][ky][kx], acc);
        acc = fmaxf(acc, 0.f);
        ov[o] = (_Float16)acc;
    }
    *(v16h*)(h1 + (size_t)idx * 16) = ov;   // NHWC, 32B aligned
}

// ---------------------------------------------------------------------------
// Kernel 2: conv2 (16 -> 16 ch, 3x3 SAME) + ReLU via v_wmma_f32_16x16x32_f16.
// WMMA tile = 16 pixels x 16 out-ch, K=32 packs two taps x 16 in-ch.
// Block = 128 thr = 4 waves over a 16x16 pixel tile (4 rows/wave).
// Interior blocks: branchless, single base + immediate-offset b128 loads.
// Border blocks: clamped address + cndmask zeroing (no saveexec ladders).
// ---------------------------------------------------------------------------
__global__ void __launch_bounds__(128)
conv2_wmma_kernel(const _Float16* __restrict__ h1, const _Float16* __restrict__ wpack,
                  const float* __restrict__ b2, _Float16* __restrict__ h2)
{
    const int lane = threadIdx.x & 31;
    const int wave = threadIdx.x >> 5;
    const int m    = lane & 15;   // A row (pixel in strip)
    const int hi   = lane >> 4;   // lane-group selector
    const int n    = m;           // B/C/D column (out channel)

    const int x0 = blockIdx.x * 16;
    const int yb = blockIdx.y * 16 + wave * 4;
    const int bb = blockIdx.z;

    // B matrices: one 32B vector load per K-group
    v16h Bm[5];
    #pragma unroll
    for (int g = 0; g < 5; ++g)
        Bm[g] = *(const v16h*)(wpack + ((size_t)g * 32 + lane) * 16);
    const float bias = b2[n];

    const _Float16* abase =
        h1 + ((((size_t)bb * HDIM + yb) * WDIM + x0 + m) << 4) + hi * 8;
    _Float16* sbase =
        h2 + ((((size_t)bb * HDIM + yb) * WDIM + x0 + 8 * hi) << 4) + n;

    const bool interior = (blockIdx.x > 0) && (blockIdx.x < gridDim.x - 1) &&
                          (blockIdx.y > 0) && (blockIdx.y < gridDim.y - 1);

    if (interior) {
        // pull the next row-group's activations toward the WGP
        __builtin_prefetch(abase + ((ptrdiff_t)5 * WDIM) * 16, 0, 1);

        #pragma unroll
        for (int r = 0; r < 4; ++r) {
            v8f acc;
            #pragma unroll
            for (int j = 0; j < 8; ++j) acc[j] = bias;

            #pragma unroll
            for (int g = 0; g < 5; ++g) {
                v16h a;
                {   // lower K half: tap 2g
                    const int T = 2 * g, dy = T / 3 - 1, dx = T % 3 - 1;
                    v8h t = *(const v8h*)(abase + ((ptrdiff_t)((r + dy) * WDIM + dx)) * 16);
                    #pragma unroll
                    for (int i = 0; i < 8; ++i) a[i] = t[i];
                }
                if (g < 4) {  // upper K half: tap 2g+1
                    const int T = 2 * g + 1, dy = T / 3 - 1, dx = T % 3 - 1;
                    v8h t = *(const v8h*)(abase + ((ptrdiff_t)((r + dy) * WDIM + dx)) * 16);
                    #pragma unroll
                    for (int i = 0; i < 8; ++i) a[8 + i] = t[i];
                } else {
                    #pragma unroll
                    for (int i = 0; i < 8; ++i) a[8 + i] = (_Float16)0.f;
                }
                acc = __builtin_amdgcn_wmma_f32_16x16x32_f16(
                    false, a, false, Bm[g], (short)0, acc, false, false);
            }
            #pragma unroll
            for (int j = 0; j < 8; ++j) {
                float v = acc[j];
                v = v > 0.f ? v : 0.f;
                sbase[((ptrdiff_t)(r * WDIM + j)) * 16] = (_Float16)v;
            }
        }
    } else {
        #pragma unroll
        for (int r = 0; r < 4; ++r) {
            const int y = yb + r;
            v8f acc;
            #pragma unroll
            for (int j = 0; j < 8; ++j) acc[j] = bias;

            #pragma unroll
            for (int g = 0; g < 5; ++g) {
                v16h a;
                {   // lower K half: tap 2g (clamped load + masked zero)
                    const int T = 2 * g, dy = T / 3 - 1, dx = T % 3 - 1;
                    int yy = y + dy, xx = x0 + dx + m;
                    bool oob = ((unsigned)yy >= HDIM) | ((unsigned)xx >= WDIM);
                    int yyc = oob ? 0 : yy, xxc = oob ? 0 : xx;
                    v8h t = *(const v8h*)(h1 + ((((size_t)bb * HDIM + yyc) * WDIM + xxc) << 4) + hi * 8);
                    #pragma unroll
                    for (int i = 0; i < 8; ++i) a[i] = oob ? (_Float16)0.f : t[i];
                }
                if (g < 4) {  // upper K half: tap 2g+1
                    const int T = 2 * g + 1, dy = T / 3 - 1, dx = T % 3 - 1;
                    int yy = y + dy, xx = x0 + dx + m;
                    bool oob = ((unsigned)yy >= HDIM) | ((unsigned)xx >= WDIM);
                    int yyc = oob ? 0 : yy, xxc = oob ? 0 : xx;
                    v8h t = *(const v8h*)(h1 + ((((size_t)bb * HDIM + yyc) * WDIM + xxc) << 4) + hi * 8);
                    #pragma unroll
                    for (int i = 0; i < 8; ++i) a[8 + i] = oob ? (_Float16)0.f : t[i];
                } else {
                    #pragma unroll
                    for (int i = 0; i < 8; ++i) a[8 + i] = (_Float16)0.f;
                }
                acc = __builtin_amdgcn_wmma_f32_16x16x32_f16(
                    false, a, false, Bm[g], (short)0, acc, false, false);
            }
            #pragma unroll
            for (int j = 0; j < 8; ++j) {
                float v = acc[j];
                v = v > 0.f ? v : 0.f;
                sbase[((ptrdiff_t)(r * WDIM + j)) * 16] = (_Float16)v;
            }
        }
    }
}

// ---------------------------------------------------------------------------
// Kernel 3: conv3 (16 -> 1 ch) -> sigmoid -> k_map; full HSV->HVI->RGB chain.
// ---------------------------------------------------------------------------
__device__ __forceinline__ float clampf(float v, float lo, float hi) {
    return fminf(fmaxf(v, lo), hi);
}

__global__ void __launch_bounds__(256)
conv3_hvi_kernel(const float* __restrict__ img, const float* __restrict__ w3,
                 const float* __restrict__ b3, const _Float16* __restrict__ h2,
                 float* __restrict__ out)
{
    const int HW = HDIM * WDIM;
    int idx = blockIdx.x * blockDim.x + threadIdx.x;
    if (idx >= BATCH * HW) return;
    int b = idx / HW;
    int p = idx - b * HW;
    int y = p >> 9;
    int x = p & (WDIM - 1);

    float acc = b3[0];
    #pragma unroll
    for (int ky = 0; ky < 3; ++ky) {
        #pragma unroll
        for (int kx = 0; kx < 3; ++kx) {
            int yy = y + ky - 1, xx = x + kx - 1;
            if (yy >= 0 && yy < HDIM && xx >= 0 && xx < WDIM) {
                const _Float16* src = h2 + ((((size_t)b * HDIM + yy) * WDIM + xx) << 4);
                v8h lo = *(const v8h*)(src);
                v8h hv = *(const v8h*)(src + 8);
                #pragma unroll
                for (int c = 0; c < 8; ++c) {
                    acc = fmaf(w3[(c * 3 + ky) * 3 + kx],       (float)lo[c], acc);
                    acc = fmaf(w3[((c + 8) * 3 + ky) * 3 + kx], (float)hv[c], acc);
                }
            }
        }
    }
    float kmap = 0.05f + 0.75f * (1.f / (1.f + expf(-acc)));

    float r  = img[(((size_t)b * 3 + 0) * HDIM + y) * WDIM + x];
    float g  = img[(((size_t)b * 3 + 1) * HDIM + y) * WDIM + x];
    float bl = img[(((size_t)b * 3 + 2) * HDIM + y) * WDIM + x];

    float value = fmaxf(r, fmaxf(g, bl));
    float vmin  = fminf(r, fminf(g, bl));
    float delta = value - vmin;

    float hue = 0.f;
    if (bl == value) hue = 4.f + (r - g) / (delta + EPS_F);
    if (g  == value) hue = 2.f + (bl - r) / (delta + EPS_F);
    if (r  == value) {
        float mm = fmodf((g - bl) / (delta + EPS_F), 6.f);
        if (mm < 0.f) mm += 6.f;          // python-style mod
        hue = mm;
    }
    if (delta == 0.f) hue = 0.f;
    hue *= (1.f / 6.f);

    float sat = (value == 0.f) ? 0.f : delta / (value + EPS_F);

    float base = sinf(value * 0.5f * PI_F) + EPS_F;
    float cs   = powf(base, kmap);
    float Hc   = cs * sat * cosf(2.f * PI_F * hue);
    float Vc   = cs * sat * sinf(2.f * PI_F * hue);
    float I    = value;
    Hc = clampf(Hc, -1.f, 1.f);
    Vc = clampf(Vc, -1.f, 1.f);
    I  = clampf(I, 0.f, 1.f);

    float cs2 = powf(sinf(I * 0.5f * PI_F) + EPS_F, kmap);
    Hc = clampf(Hc / (cs2 + EPS_F), -1.f, 1.f);
    Vc = clampf(Vc / (cs2 + EPS_F), -1.f, 1.f);

    float hh = atan2f(Vc + EPS_F, Hc + EPS_F) * (1.f / (2.f * PI_F));
    hh = hh - floorf(hh);                  // mod 1
    float ss = clampf(sqrtf(Hc * Hc + Vc * Vc + EPS_F), 0.f, 1.f);
    float vv = clampf(I, 0.f, 1.f);

    float hf  = floorf(hh * 6.f);
    float f   = hh * 6.f - hf;
    float pp  = vv * (1.f - ss);
    float qq  = vv * (1.f - f * ss);
    float tt  = vv * (1.f - (1.f - f) * ss);
    int hidx  = (int)hf;
    if (hidx < 0) hidx = 0;
    if (hidx > 5) hidx = 5;

    float rr = 0.f, gg = 0.f, bb2 = 0.f;
    switch (hidx) {
        case 0: rr = vv; gg = tt; bb2 = pp; break;
        case 1: rr = qq; gg = vv; bb2 = pp; break;
        case 2: rr = pp; gg = vv; bb2 = tt; break;
        case 3: rr = pp; gg = qq; bb2 = vv; break;
        case 4: rr = tt; gg = pp; bb2 = vv; break;
        case 5: rr = vv; gg = pp; bb2 = qq; break;
    }

    out[(((size_t)b * 3 + 0) * HDIM + y) * WDIM + x] = clampf(rr,  0.f, 1.f);
    out[(((size_t)b * 3 + 1) * HDIM + y) * WDIM + x] = clampf(gg,  0.f, 1.f);
    out[(((size_t)b * 3 + 2) * HDIM + y) * WDIM + x] = clampf(bb2, 0.f, 1.f);
}

// ---------------------------------------------------------------------------
extern "C" void kernel_launch(void* const* d_in, const int* in_sizes, int n_in,
                              void* d_out, int out_size, void* d_ws, size_t ws_size,
                              hipStream_t stream)
{
    const float* img = (const float*)d_in[0];
    const float* w1  = (const float*)d_in[1];
    const float* b1  = (const float*)d_in[2];
    const float* w2  = (const float*)d_in[3];
    const float* b2  = (const float*)d_in[4];
    const float* w3  = (const float*)d_in[5];
    const float* b3  = (const float*)d_in[6];
    float* out = (float*)d_out;

    // Workspace layout: h1 (134MB) | h2 (134MB) | packed weights (5KB)
    _Float16* h1    = (_Float16*)d_ws;
    _Float16* h2    = h1 + (size_t)BATCH * HDIM * WDIM * 16;
    _Float16* wpack = h2 + (size_t)BATCH * HDIM * WDIM * 16;

    const int npix = BATCH * HDIM * WDIM;

    pack_w2_kernel<<<1, 32, 0, stream>>>(w2, wpack);
    conv1_kernel<<<(npix + 255) / 256, 256, 0, stream>>>(img, w1, b1, h1);

    dim3 g2(WDIM / 16, HDIM / 16, BATCH);
    conv2_wmma_kernel<<<g2, 128, 0, stream>>>(h1, wpack, b2, h2);

    conv3_hvi_kernel<<<(npix + 255) / 256, 256, 0, stream>>>(img, w3, b3, h2, out);
}
